// Equi_Nonlin_Grad_Module_71554155151884
// MI455X (gfx1250) — compile-verified, matrix operand order checked by reading, not run on hardware
//
#include <hip/hip_runtime.h>

// ---------------------------------------------------------------------------
// MI455X (gfx1250, wave32) implementation of the equivariant TP + MLP fwd+bwd.
//
// Dominant work: two 1024x21504x1024 GEMMs -> v_wmma_f32_16x16x32_bf16
// (fp32 accumulate, operands staged as bf16 in ws; everything L2-resident).
// MLP GEMMs (fwd + bwd) use v_wmma_f32_16x16x4_f32 at full fp32 precision.
// Workspace requirement: ~236 MB.
// ---------------------------------------------------------------------------

typedef __attribute__((ext_vector_type(16))) __bf16 v16bf;
typedef __attribute__((ext_vector_type(8)))  float  v8f;
typedef __attribute__((ext_vector_type(2)))  float  v2f;

#define NROWS 1024
#define HIDD  1024
#define ZD    256
#define KTOT  21504   // 128*128 + 64*64 + 32*32
#define DIMT  480     // 128*1 + 64*3 + 32*5

#define ALPHA_F   0.0068193217f   // 1/sqrt(21504)
#define INV_SQ3   0.57735027f
#define INV_SQ5   0.44721360f

union Frag16 { v16bf v; uint4 q[2]; };

// ---------------------------------------------------------------------------
// S[n,k] = (ALPHA/sqrt(d)) * dot(x_u, x_v)   (bf16, k = flattened (path,u,v))
// ---------------------------------------------------------------------------
__global__ void build_S(const float* __restrict__ t, __bf16* __restrict__ S) {
  __shared__ float xs[DIMT];
  const int n = blockIdx.x, tid = threadIdx.x;
  for (int i = tid; i < DIMT; i += 256) xs[i] = t[(size_t)n * DIMT + i];
  __syncthreads();
  for (int k = tid; k < KTOT; k += 256) {
    float acc, c;
    if (k < 16384) {
      int u = k >> 7, v = k & 127;
      acc = xs[u] * xs[v]; c = ALPHA_F;
    } else if (k < 20480) {
      int kk = k - 16384, u = kk >> 6, v = kk & 63;
      const float* a = xs + 128 + u * 3;
      const float* b = xs + 128 + v * 3;
      acc = a[0]*b[0] + a[1]*b[1] + a[2]*b[2]; c = ALPHA_F * INV_SQ3;
    } else {
      int kk = k - 20480, u = kk >> 5, v = kk & 31;
      const float* a = xs + 320 + u * 5;
      const float* b = xs + 320 + v * 5;
      acc = a[0]*b[0] + a[1]*b[1] + a[2]*b[2] + a[3]*b[3] + a[4]*b[4];
      c = ALPHA_F * INV_SQ5;
    }
    S[(size_t)n * KTOT + k] = (__bf16)(acc * c);
  }
}

// ---------------------------------------------------------------------------
// WbT[w*KTOT + k] = bf16(Wf[k,w])  (LDS-tiled 32x32 transpose, coalesced I/O)
// ---------------------------------------------------------------------------
__global__ void build_WbT(const float* __restrict__ W0, const float* __restrict__ W1,
                          const float* __restrict__ W2, __bf16* __restrict__ WbT) {
  __shared__ float tile[32][33];
  const int kb = blockIdx.x * 32, wb = blockIdx.y * 32;
  const int tx = threadIdx.x & 31, ty = threadIdx.x >> 5;  // ty: 0..7
  for (int i = 0; i < 32; i += 8) {
    int k = kb + ty + i, w = wb + tx, kk = k;
    const float* src;
    if (k < 16384) src = W0;
    else if (k < 20480) { src = W1; kk = k - 16384; }
    else { src = W2; kk = k - 20480; }
    tile[ty + i][tx] = src[(size_t)kk * HIDD + w];
  }
  __syncthreads();
  for (int i = 0; i < 32; i += 8) {
    int w = wb + ty + i, k = kb + tx;
    WbT[(size_t)w * KTOT + k] = (__bf16)tile[tx][ty + i];
  }
}

// ---------------------------------------------------------------------------
// Wsym[k*HID + w] = bf16( (ALPHA/sqrt(d)) * (W[u,v,w] + W[v,u,w]) )
// ---------------------------------------------------------------------------
__global__ void build_Wsym(const float* __restrict__ W0, const float* __restrict__ W1,
                           const float* __restrict__ W2, __bf16* __restrict__ Ws) {
  size_t idx = (size_t)blockIdx.x * 256 + threadIdx.x;
  int k = (int)(idx / HIDD);
  int w = (int)(idx - (size_t)k * HIDD);
  const float* src; int u, v, mul; float c;
  if (k < 16384)      { src = W0; mul = 128; u = k >> 7;  v = k & 127; c = ALPHA_F; }
  else if (k < 20480) { src = W1; mul = 64;  int kk = k - 16384; u = kk >> 6; v = kk & 63; c = ALPHA_F * INV_SQ3; }
  else                { src = W2; mul = 32;  int kk = k - 20480; u = kk >> 5; v = kk & 31; c = ALPHA_F * INV_SQ5; }
  float a = src[((size_t)u * mul + v) * HIDD + w];
  float b = src[((size_t)v * mul + u) * HIDD + w];
  Ws[idx] = (__bf16)(c * (a + b));
}

// ---------------------------------------------------------------------------
// bf16 GEMM, "NT": A[m,k] row-major (lda=K stride), B given as N x K rows
// (ldb=K stride), C = A*B^T fp32.  Block: 256 thr = 8 waves; block tile
// 64(M) x 128(N); each wave: 16x64 (1 A-frag vs 4 B-frags, 4 WMMAs / k-step).
// ---------------------------------------------------------------------------
__global__ void gemm_bf16_nt(const __bf16* __restrict__ A, int lda,
                             const __bf16* __restrict__ B, int ldb,
                             float* __restrict__ C, long ldc, int K) {
  const int lane = threadIdx.x & 31, wave = threadIdx.x >> 5;
  const int kh = lane >> 4, r = lane & 15;
  const int mw = blockIdx.y * 64 + (wave & 3) * 16;
  const int nw = blockIdx.x * 128 + (wave >> 2) * 64;
  v8f acc0 = {}, acc1 = {}, acc2 = {}, acc3 = {};
  const __bf16* pa = A + (size_t)(mw + r) * lda + kh * 16;
  const __bf16* pb = B + (size_t)(nw + r) * ldb + kh * 16;
  const size_t bstep = (size_t)16 * ldb;
  for (int k0 = 0; k0 < K; k0 += 32) {
    Frag16 a, b0, b1, b2, b3;
    const uint4* qa = (const uint4*)(pa + k0);
    a.q[0] = qa[0]; a.q[1] = qa[1];
    const uint4* q0 = (const uint4*)(pb + k0);
    b0.q[0] = q0[0]; b0.q[1] = q0[1];
    const uint4* q1 = (const uint4*)(pb + bstep + k0);
    b1.q[0] = q1[0]; b1.q[1] = q1[1];
    const uint4* q2 = (const uint4*)(pb + 2 * bstep + k0);
    b2.q[0] = q2[0]; b2.q[1] = q2[1];
    const uint4* q3 = (const uint4*)(pb + 3 * bstep + k0);
    b3.q[0] = q3[0]; b3.q[1] = q3[1];
    acc0 = __builtin_amdgcn_wmma_f32_16x16x32_bf16(false, a.v, false, b0.v, (short)0, acc0, false, false);
    acc1 = __builtin_amdgcn_wmma_f32_16x16x32_bf16(false, a.v, false, b1.v, (short)0, acc1, false, false);
    acc2 = __builtin_amdgcn_wmma_f32_16x16x32_bf16(false, a.v, false, b2.v, (short)0, acc2, false, false);
    acc3 = __builtin_amdgcn_wmma_f32_16x16x32_bf16(false, a.v, false, b3.v, (short)0, acc3, false, false);
  }
#pragma unroll
  for (int j = 0; j < 8; ++j) {
    long row = mw + kh * 8 + j;
    C[row * ldc + nw +      r] = acc0[j];
    C[row * ldc + nw + 16 + r] = acc1[j];
    C[row * ldc + nw + 32 + r] = acc2[j];
    C[row * ldc + nw + 48 + r] = acc3[j];
  }
}

// ---------------------------------------------------------------------------
// fp32 GEMM via V_WMMA_F32_16X16X4_F32.  C[m,n] = sum_k A[m,k]*B(k,n) + bias[n]
// B element (k,n) = B[k*sK + n*sN]  (sK=N,sN=1 normal; sK=1,sN=rows for B^T).
// Block: 256 thr = 8 waves stacked along M (128 rows), 16 cols per block.
// ---------------------------------------------------------------------------
__global__ void gemm_f32(const float* __restrict__ A, int lda,
                         const float* __restrict__ B, int sK, int sN,
                         const float* __restrict__ bias,
                         float* __restrict__ C, int ldc, int K) {
  const int lane = threadIdx.x & 31, wave = threadIdx.x >> 5;
  const int kh = lane >> 4, r = lane & 15;
  const int m0 = blockIdx.y * 128 + wave * 16;
  const int n0 = blockIdx.x * 16;
  v8f acc = {};
  const float* pa = A + (size_t)(m0 + r) * lda + kh * 2;
  const float* pb = B + (size_t)(kh * 2) * sK + (size_t)(n0 + r) * sN;
  for (int k0 = 0; k0 < K; k0 += 4) {
    v2f a, b;
    a.x = pa[k0]; a.y = pa[k0 + 1];
    b.x = pb[(size_t)k0 * sK]; b.y = pb[(size_t)(k0 + 1) * sK];
    acc = __builtin_amdgcn_wmma_f32_16x16x4_f32(false, a, false, b, (short)0, acc, false, false);
  }
  const float bv = bias ? bias[n0 + r] : 0.0f;
#pragma unroll
  for (int j = 0; j < 8; ++j)
    C[(size_t)(m0 + kh * 8 + j) * ldc + n0 + r] = acc[j] + bv;
}

// ---------------------------------------------------------------------------
// LayerNorm + SiLU forward (one row per block, D = 256)
// ---------------------------------------------------------------------------
__global__ void ln_silu_fwd(const float* __restrict__ a, const float* __restrict__ g,
                            const float* __restrict__ be, float* __restrict__ h,
                            float* __restrict__ mu_o, float* __restrict__ iv_o) {
  __shared__ float r1[256], r2[256];
  const int n = blockIdx.x, tid = threadIdx.x;
  float v = a[(size_t)n * ZD + tid];
  r1[tid] = v; __syncthreads();
  for (int s = 128; s > 0; s >>= 1) { if (tid < s) r1[tid] += r1[tid + s]; __syncthreads(); }
  float mu = r1[0] * (1.0f / ZD);
  float d = v - mu;
  r2[tid] = d * d; __syncthreads();
  for (int s = 128; s > 0; s >>= 1) { if (tid < s) r2[tid] += r2[tid + s]; __syncthreads(); }
  float iv = rsqrtf(r2[0] * (1.0f / ZD) + 1e-6f);
  float l = d * iv * g[tid] + be[tid];
  float sig = 1.0f / (1.0f + __expf(-l));
  h[(size_t)n * ZD + tid] = l * sig;
  if (tid == 0) { mu_o[n] = mu; iv_o[n] = iv; }
}

// ---------------------------------------------------------------------------
// Backward through SiLU + LayerNorm.  dh_stride==0 => dh is a broadcast vector.
// ---------------------------------------------------------------------------
__global__ void ln_silu_bwd(const float* __restrict__ dh, int dh_stride,
                            const float* __restrict__ a,
                            const float* __restrict__ mu_i, const float* __restrict__ iv_i,
                            const float* __restrict__ g, const float* __restrict__ be,
                            float* __restrict__ da) {
  __shared__ float r1[256], r2[256];
  const int n = blockIdx.x, tid = threadIdx.x;
  const float mu = mu_i[n], iv = iv_i[n];
  float v  = a[(size_t)n * ZD + tid];
  float nh = (v - mu) * iv;
  float l  = nh * g[tid] + be[tid];
  float sig = 1.0f / (1.0f + __expf(-l));
  float dsl = sig * (1.0f + l * (1.0f - sig));      // d silu / d l
  float dhv = dh[(size_t)n * dh_stride + tid];
  float dl  = dhv * dsl;
  float dn  = dl * g[tid];
  r1[tid] = dn; r2[tid] = dn * nh; __syncthreads();
  for (int s = 128; s > 0; s >>= 1) {
    if (tid < s) { r1[tid] += r1[tid + s]; r2[tid] += r2[tid + s]; }
    __syncthreads();
  }
  float m1 = r1[0] * (1.0f / ZD), m2 = r2[0] * (1.0f / ZD);
  da[(size_t)n * ZD + tid] = iv * (dn - m1 - nh * m2);
}

// dh3[j] = sum_k w4[j,k]  (upstream grad is all-ones)
__global__ void rowsum_w4(const float* __restrict__ w4, float* __restrict__ c3) {
  int j = threadIdx.x;
  float s = 0.f;
  for (int k = 0; k < ZD; ++k) s += w4[(size_t)j * ZD + k];
  c3[j] = s;
}

__global__ void f32_to_bf16(const float* __restrict__ src, __bf16* __restrict__ dst, int n) {
  int i = blockIdx.x * 256 + threadIdx.x;
  if (i < n) dst[i] = (__bf16)src[i];
}

// ---------------------------------------------------------------------------
// dt[n, (path,u,m)] = sum_v dSsym[n, base + u*mul + v] * x[n, xoff + v*d + m]
// ---------------------------------------------------------------------------
__global__ void tp_bwd_dt(const float* __restrict__ t, const float* __restrict__ dS,
                          float* __restrict__ y) {
  __shared__ float xs[DIMT];
  const int n = blockIdx.x, tid = threadIdx.x;
  for (int i = tid; i < DIMT; i += 256) xs[i] = t[(size_t)n * DIMT + i];
  __syncthreads();
  const float* ds = dS + (size_t)n * KTOT;
  for (int o = tid; o < DIMT; o += 256) {
    int u, m, mul, d, base, xoff;
    if (o < 128)      { u = o; m = 0; mul = 128; d = 1; base = 0;     xoff = 0;   }
    else if (o < 320) { int oo = o - 128; u = oo / 3; m = oo - u * 3; mul = 64; d = 3; base = 16384; xoff = 128; }
    else              { int oo = o - 320; u = oo / 5; m = oo - u * 5; mul = 32; d = 5; base = 20480; xoff = 320; }
    const float* dsr = ds + base + u * mul;
    const float* xr  = xs + xoff + m;
    float acc = 0.f;
    for (int v = 0; v < mul; ++v) acc += dsr[v] * xr[v * d];
    y[(size_t)n * DIMT + o] = acc;
  }
}

// ---------------------------------------------------------------------------
extern "C" void kernel_launch(void* const* d_in, const int* in_sizes, int n_in,
                              void* d_out, int out_size, void* d_ws, size_t ws_size,
                              hipStream_t stream) {
  (void)in_sizes; (void)n_in; (void)out_size; (void)ws_size;
  const float* t   = (const float*)d_in[0];
  const float* W0  = (const float*)d_in[1];
  const float* W1  = (const float*)d_in[2];
  const float* W2  = (const float*)d_in[3];
  const float* w1  = (const float*)d_in[4];
  const float* b1  = (const float*)d_in[5];
  const float* g1  = (const float*)d_in[6];
  const float* be1 = (const float*)d_in[7];
  const float* w2  = (const float*)d_in[8];
  const float* b2  = (const float*)d_in[9];
  const float* g2  = (const float*)d_in[10];
  const float* be2 = (const float*)d_in[11];
  const float* w3  = (const float*)d_in[12];
  const float* b3  = (const float*)d_in[13];
  const float* g3  = (const float*)d_in[14];
  const float* be3 = (const float*)d_in[15];
  const float* w4  = (const float*)d_in[16];
  const float* b4  = (const float*)d_in[17];

  float* out_x = (float*)d_out;                   // N x 256
  float* out_y = out_x + (size_t)NROWS * ZD;      // N x 480

  // ---- workspace carve-up (all 256B aligned) ----
  char* p = (char*)d_ws;
  auto alloc = [&](size_t bytes) { char* r = p; p += (bytes + 255) & ~(size_t)255; return r; };
  __bf16* WbT   = (__bf16*)alloc((size_t)KTOT * HIDD * 2);   // 44 MB  fwd B
  __bf16* Wsym  = (__bf16*)alloc((size_t)KTOT * HIDD * 2);   // 44 MB  bwd B
  __bf16* S     = (__bf16*)alloc((size_t)NROWS * KTOT * 2);  // 44 MB
  float*  z     = (float*) alloc((size_t)NROWS * HIDD * 4);  // 4 MB
  float*  a1    = (float*) alloc((size_t)NROWS * ZD * 4);
  float*  a2    = (float*) alloc((size_t)NROWS * ZD * 4);
  float*  a3    = (float*) alloc((size_t)NROWS * ZD * 4);
  float*  h1    = (float*) alloc((size_t)NROWS * ZD * 4);
  float*  h2    = (float*) alloc((size_t)NROWS * ZD * 4);
  float*  h3    = (float*) alloc((size_t)NROWS * ZD * 4);
  float*  mu1   = (float*) alloc(NROWS * 4);
  float*  iv1   = (float*) alloc(NROWS * 4);
  float*  mu2   = (float*) alloc(NROWS * 4);
  float*  iv2   = (float*) alloc(NROWS * 4);
  float*  mu3   = (float*) alloc(NROWS * 4);
  float*  iv3   = (float*) alloc(NROWS * 4);
  float*  c3    = (float*) alloc(ZD * 4);
  float*  dbA   = (float*) alloc((size_t)NROWS * ZD * 4);
  float*  dbB   = (float*) alloc((size_t)NROWS * ZD * 4);
  float*  dz    = (float*) alloc((size_t)NROWS * HIDD * 4);  // 4 MB
  __bf16* dzb   = (__bf16*)alloc((size_t)NROWS * HIDD * 2);  // 2 MB
  float*  dSsym = (float*) alloc((size_t)NROWS * KTOT * 4);  // 88 MB

  // ---- stage bf16 operands ----
  build_WbT <<<dim3(KTOT / 32, HIDD / 32), 256, 0, stream>>>(W0, W1, W2, WbT);
  build_Wsym<<<(unsigned)((size_t)KTOT * HIDD / 256), 256, 0, stream>>>(W0, W1, W2, Wsym);
  build_S   <<<NROWS, 256, 0, stream>>>(t, S);

  // ---- forward: z = S @ Wf (bf16 WMMA, fp32 accum) ----
  gemm_bf16_nt<<<dim3(HIDD / 128, NROWS / 64), 256, 0, stream>>>(S, KTOT, WbT, KTOT, z, HIDD, KTOT);

  // ---- forward MLP (fp32 WMMA) ----
  gemm_f32<<<dim3(ZD / 16, NROWS / 128), 256, 0, stream>>>(z,  HIDD, w1, ZD, 1, b1, a1, ZD, HIDD);
  ln_silu_fwd<<<NROWS, 256, 0, stream>>>(a1, g1, be1, h1, mu1, iv1);
  gemm_f32<<<dim3(ZD / 16, NROWS / 128), 256, 0, stream>>>(h1, ZD,   w2, ZD, 1, b2, a2, ZD, ZD);
  ln_silu_fwd<<<NROWS, 256, 0, stream>>>(a2, g2, be2, h2, mu2, iv2);
  gemm_f32<<<dim3(ZD / 16, NROWS / 128), 256, 0, stream>>>(h2, ZD,   w3, ZD, 1, b3, a3, ZD, ZD);
  ln_silu_fwd<<<NROWS, 256, 0, stream>>>(a3, g3, be3, h3, mu3, iv3);
  gemm_f32<<<dim3(ZD / 16, NROWS / 128), 256, 0, stream>>>(h3, ZD,   w4, ZD, 1, b4, out_x, ZD, ZD);

  // ---- backward MLP ----
  rowsum_w4<<<1, 256, 0, stream>>>(w4, c3);                                 // dh3 (broadcast)
  ln_silu_bwd<<<NROWS, 256, 0, stream>>>(c3, 0, a3, mu3, iv3, g3, be3, dbA);      // da3
  gemm_f32<<<dim3(ZD / 16, NROWS / 128), 256, 0, stream>>>(dbA, ZD, w3, 1, ZD, nullptr, dbB, ZD, ZD); // dh2
  ln_silu_bwd<<<NROWS, 256, 0, stream>>>(dbB, ZD, a2, mu2, iv2, g2, be2, dbA);    // da2
  gemm_f32<<<dim3(ZD / 16, NROWS / 128), 256, 0, stream>>>(dbA, ZD, w2, 1, ZD, nullptr, dbB, ZD, ZD); // dh1
  ln_silu_bwd<<<NROWS, 256, 0, stream>>>(dbB, ZD, a1, mu1, iv1, g1, be1, dbA);    // da1
  gemm_f32<<<dim3(HIDD / 16, NROWS / 128), 256, 0, stream>>>(dbA, ZD, w1, 1, ZD, nullptr, dz, HIDD, ZD); // dz

  // ---- backward TP: dSsym = dz @ Wsym^T, then contract against x ----
  f32_to_bf16<<<(NROWS * HIDD + 255) / 256, 256, 0, stream>>>(dz, dzb, NROWS * HIDD);
  gemm_bf16_nt<<<dim3(KTOT / 128, NROWS / 64), 256, 0, stream>>>(dzb, HIDD, Wsym, HIDD, dSsym, KTOT, HIDD);
  tp_bwd_dt<<<NROWS, 256, 0, stream>>>(t, dSsym, out_y);
}
